// BehlerParrinelloNN_46591805227419
// MI455X (gfx1250) — compile-verified, matrix operand order checked by reading, not run on hardware
//
#include <hip/hip_runtime.h>
#include <math.h>

typedef __attribute__((ext_vector_type(16))) _Float16 v16h;
typedef __attribute__((ext_vector_type(2)))  __fp16   v2fp16;
typedef __attribute__((ext_vector_type(8)))  float    v8f;

union FragAB {
    v16h h;
    unsigned int u[8];
    uint4 q[2];
};

// Single-instruction packed f32->2xf16 convert (v_cvt_pk_rtz_f16_f32).
__device__ __forceinline__ unsigned int pack2h(float lo, float hi) {
#if defined(__has_builtin) && __has_builtin(__builtin_amdgcn_cvt_pkrtz)
    union { v2fp16 h; unsigned int u; } p;
    p.h = __builtin_amdgcn_cvt_pkrtz(lo, hi);
    return p.u;
#else
    union { _Float16 h[2]; unsigned int u; } p;
    p.h[0] = (_Float16)lo;
    p.h[1] = (_Float16)hi;
    return p.u;
#endif
}

// CDNA5 native V_TANH_F32 TRANS op (co-executes with WMMA pipe).
__device__ __forceinline__ float fast_tanh(float x) {
#if defined(__has_builtin)
#if __has_builtin(__builtin_amdgcn_tanhf)
    return __builtin_amdgcn_tanhf(x);
#elif __has_builtin(__builtin_amdgcn_tanh_f32)
    return __builtin_amdgcn_tanh_f32(x);
#else
    float e = __expf(2.0f * fabsf(x));
    float t = 1.0f - 2.0f / (e + 1.0f);
    return copysignf(t, x);
#endif
#else
    float e = __expf(2.0f * fabsf(x));
    float t = 1.0f - 2.0f / (e + 1.0f);
    return copysignf(t, x);
#endif
}

#define BLOCK 256
#define WPB   8     // wave32s per block
#define APB   128   // atoms per block (16 per wave)

// ISA K-pair start for 16-bit A/B fragment VGPR v, lane-half h.
__device__ __forceinline__ int fragK(int v, int h) {
    return (v & 3) * 2 + ((v & 4) ? 16 : 0) + h * 8;
}

// One wave processes 16 atoms through the whole MLP with f16 WMMA, f32 accum.
__global__ __launch_bounds__(BLOCK) void bp_mlp_kernel(
    const float* __restrict__ feats,   // [n_atoms][128] f32
    const int*   __restrict__ midx,    // [n_atoms]
    const float* __restrict__ W1,      // [128][64]
    const float* __restrict__ B1,      // [64]
    const float* __restrict__ W2,      // [64][32]
    const float* __restrict__ B2,      // [32]
    const float* __restrict__ W3,      // [32][1]
    const float* __restrict__ B3,      // [1]
    float*       __restrict__ molE,    // [n_molecules]
    int n_atoms)
{
    // Weights stored in LDS in *fragment order*: the 8 dwords lane L needs for
    // WMMA B operand VGPRs v0..v7 are contiguous -> 2x ds_load_b128 per frag,
    // loaded straight into the WMMA source registers (no v_mov gathering).
    __shared__ __align__(16) unsigned int sW1f[4 * 4 * 32 * 8]; // [nt][ks][lane][v] 16 KB
    __shared__ __align__(16) unsigned int sW2f[2 * 2 * 32 * 8]; // [nt][ks][lane][v]  4 KB
    __shared__ float sb1[64], sb2[32], sw3[32], sb3[1];
    __shared__ _Float16 sTile[WPB][16 * 128]; // feats tile, h1 aliases it (32 KB)
    __shared__ float sE[WPB][16];             // per-wave atomic energies

    const int tid  = threadIdx.x;
    const int wave = tid >> 5;
    const int lane = tid & 31;
    const int idx  = lane & 15;   // M for A / N for B / N for C-D
    const int half = lane >> 4;

    // ---- Stage weights pre-permuted into fragment order ----
    for (int i = tid; i < 4 * 4 * 32 * 8; i += BLOCK) {
        int v  = i & 7;
        int ln = (i >> 3) & 31;
        int ks = (i >> 8) & 3;
        int nt = i >> 10;
        int K  = 32 * ks + fragK(v, ln >> 4);
        int N  = nt * 16 + (ln & 15);
        sW1f[i] = pack2h(W1[K * 64 + N], W1[(K + 1) * 64 + N]);
    }
    for (int i = tid; i < 2 * 2 * 32 * 8; i += BLOCK) {
        int v  = i & 7;
        int ln = (i >> 3) & 31;
        int ks = (i >> 8) & 1;
        int nt = i >> 9;
        int K  = 32 * ks + fragK(v, ln >> 4);
        int N  = nt * 16 + (ln & 15);
        sW2f[i] = pack2h(W2[K * 32 + N], W2[(K + 1) * 32 + N]);
    }
    if (tid < 64) sb1[tid] = B1[tid];
    if (tid < 32) sb2[tid] = B2[tid];
    if (tid < 32) sw3[tid] = W3[tid];
    if (tid == 0) sb3[0] = B3[0];

    // ---- Stage 16x128 feature tile: coalesced float4 loads, convert to f16 ----
    const int atom0 = blockIdx.x * APB + wave * 16;
    if (atom0 + 16 <= n_atoms) {
#pragma unroll 4
        for (int r = 0; r < 16; ++r) {
            float4 f = *(const float4*)(feats + (size_t)(atom0 + r) * 128 + lane * 4);
            uint2 p;
            p.x = pack2h(f.x, f.y);
            p.y = pack2h(f.z, f.w);
            *(uint2*)(&sTile[wave][r * 128 + lane * 4]) = p;
        }
    } else {
        for (int r = 0; r < 16; ++r) {
            int atom = atom0 + r;
            float4 f = make_float4(0.f, 0.f, 0.f, 0.f);
            if (atom < n_atoms)
                f = *(const float4*)(feats + (size_t)atom * 128 + lane * 4);
            uint2 p;
            p.x = pack2h(f.x, f.y);
            p.y = pack2h(f.z, f.w);
            *(uint2*)(&sTile[wave][r * 128 + lane * 4]) = p;
        }
    }
    __syncthreads();

    if (lane < 16) sE[wave][lane] = sb3[0];  // start energies at b3

    // ---- Prebuild layer-1 A fragments (K pairs contiguous -> ds_load_b128) ----
    FragAB A1[4];
#pragma unroll
    for (int ks = 0; ks < 4; ++ks) {
        const _Float16* row = &sTile[wave][idx * 128 + 32 * ks + half * 8];
        A1[ks].q[0] = *(const uint4*)(row);        // K = base+0..7   (v0..v3)
        A1[ks].q[1] = *(const uint4*)(row + 16);   // K = base+16..23 (v4..v7)
    }

    // ---- Layer 1: [16x128] @ [128x64], 4 N-tiles x 4 K-steps of WMMA ----
#pragma unroll
    for (int nt = 0; nt < 4; ++nt) {
        v8f acc = {0.f, 0.f, 0.f, 0.f, 0.f, 0.f, 0.f, 0.f};
#pragma unroll
        for (int ks = 0; ks < 4; ++ks) {
            FragAB Bf;
            const uint4* src = (const uint4*)&sW1f[((nt * 4 + ks) * 32 + lane) * 8];
            Bf.q[0] = src[0];
            Bf.q[1] = src[1];
            acc = __builtin_amdgcn_wmma_f32_16x16x32_f16(
                false, A1[ks].h, false, Bf.h, (short)0, acc, false, false);
        }
        int N = nt * 16 + idx;
        float b = sb1[N];
#pragma unroll
        for (int r = 0; r < 8; ++r) {
            int M = r + 8 * half;  // C/D layout: VGPR r, lane-half selects M block
            sTile[wave][M * 64 + N] = (_Float16)fast_tanh(acc[r] + b);
        }
    }

    // ---- Layer 2: [16x64] @ [64x32], 2 N-tiles x 2 K-steps; fuse layer 3 ----
#pragma unroll
    for (int nt = 0; nt < 2; ++nt) {
        v8f acc = {0.f, 0.f, 0.f, 0.f, 0.f, 0.f, 0.f, 0.f};
#pragma unroll
        for (int ks = 0; ks < 2; ++ks) {
            FragAB Af, Bf;
            const _Float16* row = &sTile[wave][idx * 64 + 32 * ks + half * 8];
            Af.q[0] = *(const uint4*)(row);
            Af.q[1] = *(const uint4*)(row + 16);
            const uint4* src = (const uint4*)&sW2f[((nt * 2 + ks) * 32 + lane) * 8];
            Bf.q[0] = src[0];
            Bf.q[1] = src[1];
            acc = __builtin_amdgcn_wmma_f32_16x16x32_f16(
                false, Af.h, false, Bf.h, (short)0, acc, false, false);
        }
        // Layer 3 (32 -> 1): e[M] += tanh(h2[M][N]) * w3[N], reduce via ds_add_f32
        int N = nt * 16 + idx;
        float b = sb2[N];
        float w = sw3[N];
#pragma unroll
        for (int r = 0; r < 8; ++r) {
            int M = r + 8 * half;
            atomicAdd(&sE[wave][M], fast_tanh(acc[r] + b) * w);
        }
    }

    // ---- Scatter atomic energies into molecule bins ----
    if (lane < 16) {
        int atom = atom0 + lane;
        if (atom < n_atoms)
            atomicAdd(&molE[midx[atom]], sE[wave][lane]);
    }
}

__global__ void bp_zero_kernel(float* __restrict__ p, int n) {
    int i = blockIdx.x * blockDim.x + threadIdx.x;
    if (i < n) p[i] = 0.f;
}

extern "C" void kernel_launch(void* const* d_in, const int* in_sizes, int n_in,
                              void* d_out, int out_size, void* d_ws, size_t ws_size,
                              hipStream_t stream) {
    (void)n_in; (void)d_ws; (void)ws_size;
    const float* featsH = (const float*)d_in[0];
    const float* featsO = (const float*)d_in[1];
    const int*   midxH  = (const int*)d_in[2];
    const int*   midxO  = (const int*)d_in[3];
    // d_in[4] = n_molecules scalar on device; use out_size on host instead.
    const float* W1H = (const float*)d_in[5];
    const float* b1H = (const float*)d_in[6];
    const float* W2H = (const float*)d_in[7];
    const float* b2H = (const float*)d_in[8];
    const float* W3H = (const float*)d_in[9];
    const float* b3H = (const float*)d_in[10];
    const float* W1O = (const float*)d_in[11];
    const float* b1O = (const float*)d_in[12];
    const float* W2O = (const float*)d_in[13];
    const float* b2O = (const float*)d_in[14];
    const float* W3O = (const float*)d_in[15];
    const float* b3O = (const float*)d_in[16];
    float* molE = (float*)d_out;

    const int nH = in_sizes[0] / 128;
    const int nO = in_sizes[1] / 128;

    bp_zero_kernel<<<(out_size + 255) / 256, 256, 0, stream>>>(molE, out_size);

    const int gH = (nH + APB - 1) / APB;
    const int gO = (nO + APB - 1) / APB;
    bp_mlp_kernel<<<gH, BLOCK, 0, stream>>>(featsH, midxH, W1H, b1H, W2H, b2H,
                                            W3H, b3H, molE, nH);
    bp_mlp_kernel<<<gO, BLOCK, 0, stream>>>(featsO, midxO, W1O, b1O, W2O, b2O,
                                            W3O, b3O, molE, nO);
}
